// BertEncoder_23055384445050
// MI455X (gfx1250) — matrix operand order repaired and checked by reading, not compiled, not run on hardware
//
#include <hip/hip_runtime.h>
#include <hip/hip_bf16.h>

// Problem constants (match the reference).
#define BB 32
#define LL 2048
#define HH 768
#define DD 32
#define CC 64
#define NEGV (-900000.0f)

typedef __attribute__((ext_vector_type(16))) __bf16 v16bf;
typedef __attribute__((ext_vector_type(8)))  float  v8f;
typedef __attribute__((ext_vector_type(8)))  int    v8i;

// Round-to-nearest-even f32 -> bf16 bit pattern (off hot path).
__device__ __forceinline__ unsigned short bf16_bits_rne(float f) {
    unsigned u = __builtin_bit_cast(unsigned, f);
    unsigned r = u + 0x7FFFu + ((u >> 16) & 1u);
    return (unsigned short)(r >> 16);
}

// Pack two f32 into a bf16 pair (round-half-up + v_perm_b32). Hot path: 3 VALU.
__device__ __forceinline__ unsigned pack_bf16_pair(float f0, float f1) {
    unsigned u0 = __builtin_bit_cast(unsigned, f0) + 0x8000u;
    unsigned u1 = __builtin_bit_cast(unsigned, f1) + 0x8000u;
    // result = { u1[31:16], u0[31:16] }
    return __builtin_amdgcn_perm(u1, u0, 0x07060302u);
}

// ---------------------------------------------------------------------------
// Kernel 1: per-clause token scores + masked softmax -> weights in d_ws.
// One block per (b, d) clause; 8 waves x 8 tokens; float4 coalesced dot
// (6x global_load_b128 per token per lane) + wave32 shfl_xor reduce.
// ---------------------------------------------------------------------------
__global__ void k_alpha(const float* __restrict__ hs,
                        const float* __restrict__ fc_w,
                        const float* __restrict__ fc_b,
                        const int*   __restrict__ clause_len,
                        const int*   __restrict__ doc_len,
                        float*       __restrict__ w_out) {
    __shared__ __align__(16) float s_fcw[HH];
    __shared__ float s_alpha[CC];
    __shared__ int   s_off, s_len;

    const int bd  = blockIdx.x;          // 0..1023
    const int b   = bd >> 5;
    const int d   = bd & 31;
    const int tid = threadIdx.x;

    for (int i = tid; i < HH; i += 256) s_fcw[i] = fc_w[i];
    if (tid == 0) {
        int off = 0;
        for (int j = 0; j < d; ++j) off += clause_len[b * DD + j];
        s_off = off;
        s_len = clause_len[b * DD + d];
    }
    __syncthreads();

    const int off  = s_off;
    const int len  = s_len;
    const int wave = tid >> 5;
    const int lane = tid & 31;
    const float4* w4 = (const float4*)s_fcw;   // HH/4 = 192 float4

    // 8 waves x 8 tokens each = 64 token slots.
    for (int i = 0; i < 8; ++i) {
        const int c = wave * 8 + i;
        float partial = 0.f;
        if (c < len) {
            const float4* row4 =
                (const float4*)(hs + ((size_t)b * LL + (size_t)(off + c)) * HH);
            #pragma unroll
            for (int q = 0; q < 6; ++q) {
                const float4 a = row4[lane + 32 * q];
                const float4 w = w4[lane + 32 * q];
                partial += a.x * w.x + a.y * w.y + a.z * w.z + a.w * w.w;
            }
        }
        #pragma unroll
        for (int s = 16; s > 0; s >>= 1) partial += __shfl_xor(partial, s, 32);
        if (lane == 0) s_alpha[c] = (c < len) ? (partial + fc_b[0]) : NEGV;
    }
    __syncthreads();

    if (tid == 0) {
        float m = -3.4e38f;
        for (int c = 0; c < CC; ++c) m = fmaxf(m, s_alpha[c]);
        float z = 0.f;
        for (int c = 0; c < CC; ++c) z += __expf(s_alpha[c] - m);
        const bool valid = d < doc_len[b];
        const float inv = valid ? (1.f / z) : 0.f;   // invalid doc rows pool to 0
        float* wo = w_out + (size_t)bd * CC;
        for (int c = 0; c < CC; ++c) wo[c] = __expf(s_alpha[c] - m) * inv;
    }
}

// ---------------------------------------------------------------------------
// Kernel 2: attention pooling as a WMMA GEMM.
// out[16 clauses, 768] = A[16, K] x hs[tok0 : tok0+K, 768], K-chunked by 32.
// A (sparse softmax-weight matrix, one nonzero per token column) is prebuilt
// per lane per chunk into LDS, so the hot loop is branchless:
//   2x ds_load_b128 (A frag) + 16x global_load_b32 (B) + pack + 1x WMMA.
// Padded K rows stay inside hs[b] and have zero A weight, so no tail code.
// grid = (6 n-chunks of 128 cols, 2 clause tiles, 32 batches), 256 thr.
// ---------------------------------------------------------------------------
__global__ void k_pool(const float* __restrict__ hs,
                       const int*   __restrict__ clause_len,
                       const float* __restrict__ w_in,
                       float*       __restrict__ out) {
    __shared__ int s_start[16];
    __shared__ int s_len[16];
    __shared__ int s_tok0, s_tokEnd;
    // Packed token table: (ownerRow << 16) | bf16(weight); 0xFFFF0000 = unused.
    __shared__ unsigned s_tok[1024];
    // Prebuilt A fragments: [chunk][lane][8 dwords] (16 bf16 per lane).
    __shared__ __align__(16) unsigned s_afrag[32 * 32 * 8];   // 32 KB

    const int nchunk = blockIdx.x;       // 0..5  (128 columns each)
    const int dt     = blockIdx.y;       // 0..1  (16 clauses each)
    const int b      = blockIdx.z;       // 0..31
    const int d0     = dt * 16;
    const int tid    = threadIdx.x;
    const int wave   = tid >> 5;
    const int lane   = tid & 31;
    const int hi     = lane >> 4;        // lane half (0 | 1)
    const int M      = lane & 15;        // A row owned by this lane

    // Phase 1: clause offsets + clear token table.
    if (tid == 0) {
        int off = 0;
        for (int j = 0; j < d0; ++j) off += clause_len[b * DD + j];
        s_tok0 = off;
        for (int r = 0; r < 16; ++r) {
            const int cl = clause_len[b * DD + d0 + r];
            s_start[r] = off;
            s_len[r]   = cl;
            off += cl;
        }
        s_tokEnd = off;
    }
    for (int i = tid; i < 1024; i += 256) s_tok[i] = 0xFFFF0000u;
    __syncthreads();

    // Phase 2: scatter softmax weights into the token table.
    const int tok0 = s_tok0;
    for (int s = tid; s < 16 * CC; s += 256) {
        const int r = s >> 6;
        const int c = s & 63;
        if (c < s_len[r]) {
            const float wv = w_in[((size_t)(b * DD + d0 + r)) * CC + c];
            s_tok[(s_start[r] - tok0) + c] =
                ((unsigned)r << 16) | (unsigned)bf16_bits_rne(wv);
        }
    }
    __syncthreads();

    // Phase 3: prebuild per-lane A fragments for every K-chunk.
    // A k order used here (contiguous, same as B): dword p holds K = 16*hi + 2p(+1).
    const int span    = s_tokEnd - tok0;
    const int nChunks = (span + 31) >> 5;
    const unsigned Mu = (unsigned)M;
    for (int ch = wave; ch < nChunks; ch += 8) {
        const int kk = ch << 5;
        unsigned* dst = &s_afrag[((ch << 5) + lane) << 3];
        #pragma unroll
        for (int p = 0; p < 8; ++p) {
            const int base = kk + (hi << 4) + 2 * p;
            const unsigned u0 = s_tok[base];
            const unsigned u1 = s_tok[base + 1];
            const unsigned lo = ((u0 >> 16) == Mu) ? (u0 & 0xFFFFu) : 0u;
            const unsigned hh = ((u1 >> 16) == Mu) ? (u1 << 16)     : 0u;
            dst[p] = lo | hh;
        }
    }
    __syncthreads();

    // Phase 4: branchless WMMA K-loop (unrolled x2 to deepen the load pipe).
    const int ncol0  = nchunk * 128 + wave * 16;  // wave's 16-col N tile
    const int colB   = ncol0 + (lane & 15);       // B column owned by this lane
    const float* hsb = hs + (size_t)b * LL * HH;

    v8f acc = {0.f, 0.f, 0.f, 0.f, 0.f, 0.f, 0.f, 0.f};

    #pragma unroll 2
    for (int ch = 0; ch < nChunks; ++ch) {
        const int kk = ch << 5;
        // Speculative prefetch of the next K-chunk (global_prefetch_b8).
        __builtin_prefetch(hsb + (size_t)(tok0 + kk + 32) * HH + colB, 0, 1);

        // A fragment: two aligned 16B LDS loads.
        const int4 fa0 = *(const int4*)&s_afrag[((ch << 5) + lane) << 3];
        const int4 fa1 = *(const int4*)(&s_afrag[(((ch << 5) + lane) << 3) + 4]);
        v8i ai;
        ai[0] = fa0.x; ai[1] = fa0.y; ai[2] = fa0.z; ai[3] = fa0.w;
        ai[4] = fa1.x; ai[5] = fa1.y; ai[6] = fa1.z; ai[7] = fa1.w;
        const v16bf Af = __builtin_bit_cast(v16bf, ai);

        // B fragment: 16 unconditional global loads (immediate row offsets),
        // contiguous k order (same as A): lane<16 -> K = e, lane>=16 -> K = 16 + e.
        const float* bp = hsb + (size_t)(tok0 + kk + (hi << 4)) * HH + colB;
        v8i bi;
        #pragma unroll
        for (int p = 0; p < 8; ++p)
            bi[p] = (int)pack_bf16_pair(bp[(size_t)(2 * p) * HH],
                                        bp[(size_t)(2 * p + 1) * HH]);
        const v16bf Bf = __builtin_bit_cast(v16bf, bi);

        acc = __builtin_amdgcn_wmma_f32_16x16x32_bf16(
            /*neg_a=*/false, Af, /*neg_b=*/false, Bf,
            /*c_mod=*/(short)0, acc, /*reuse_a=*/false, /*reuse_b=*/false);
    }

    // C/D layout: VGPR j -> row (j + 8*hi), col = lane&15.
    #pragma unroll
    for (int j = 0; j < 8; ++j) {
        const int row = j + (hi << 3);
        out[((size_t)(b * DD + d0 + row)) * HH + (size_t)(ncol0 + (lane & 15))] = acc[j];
    }
}

extern "C" void kernel_launch(void* const* d_in, const int* in_sizes, int n_in,
                              void* d_out, int out_size, void* d_ws, size_t ws_size,
                              hipStream_t stream) {
    const float* hs         = (const float*)d_in[0];
    const float* fc_w       = (const float*)d_in[1];
    const float* fc_b       = (const float*)d_in[2];
    const int*   clause_len = (const int*)d_in[3];
    const int*   doc_len    = (const int*)d_in[4];
    float*       out        = (float*)d_out;
    float*       wbuf       = (float*)d_ws;   // B*D*C floats = 256 KB

    k_alpha<<<dim3(BB * DD), dim3(256), 0, stream>>>(hs, fc_w, fc_b, clause_len, doc_len, wbuf);
    k_pool <<<dim3(6, 2, BB), dim3(256), 0, stream>>>(hs, clause_len, wbuf, out);
}
